// DeepAlignmentLoss_45887430591178
// MI455X (gfx1250) — matrix operand order repaired and checked by reading, not compiled, not run on hardware
//
#include <hip/hip_runtime.h>

// ---------------------------------------------------------------------------
// DeepAlignmentLoss on MI455X (gfx1250):
//   C = 1 - cos_sim(X, Y)   (bf16 WMMA, f32 accumulate, one WG per batch)
//   IPOT closed form: T_k = diag(u) * exp(-k*C/beta) * diag(v)
//   loss_b = -sum_ij u_i exp(-2*ITERS*C_ij) v_j C_ij ; out = mean_sp + mean_tm
// ---------------------------------------------------------------------------

typedef __attribute__((ext_vector_type(16))) __bf16 v16bf;
typedef __attribute__((ext_vector_type(8)))  __bf16 v8bf;
typedef __attribute__((ext_vector_type(8)))  float  v8f;

#define D_DIM 768
#define KC    32            // K per WMMA step (bf16)
#define AST   40            // LDS staging row stride in bf16 elems (80B, 16B aligned)
#define IPOT_ITERS 20
// beta = 0.5  ->  exp(-k*C/beta) = exp(-2k*C)

// A-fragment (16x32 bf16, MxK): lane = (r, h); K in {h*8..h*8+7} U {16+h*8..23+h*8}
__device__ __forceinline__ v16bf load_fragA(const __bf16* buf, int tile, int lane) {
    const int r  = lane & 15;
    const int hl = lane >> 4;
    const __bf16* p = buf + (tile * 16 + r) * AST + hl * 8;
    union { v16bf v; v8bf h[2]; } u;
    u.h[0] = *(const v8bf*)(p);        // ds_load_b128
    u.h[1] = *(const v8bf*)(p + 16);   // ds_load_b128
    return u.v;
}

// B-fragment (32x16 bf16, KxN): lane = (n, h); K in {h*8..h*8+7} U {16+h*8..23+h*8}
__device__ __forceinline__ v16bf load_fragB(const __bf16* buf, int tile, int lane) {
    const int n  = lane & 15;
    const int hl = lane >> 4;
    const __bf16* p = buf + (tile * 16 + n) * AST + hl * 8;
    union { v16bf v; v8bf h[2]; } u;
    u.h[0] = *(const v8bf*)(p);
    u.h[1] = *(const v8bf*)(p + 16);
    return u.v;
}

template<int N>
__global__ __launch_bounds__(((N + 15) / 16) * 32)
void align_ipot_kernel(const float* __restrict__ X,
                       const float* __restrict__ Y,
                       float* __restrict__ partial) {
    constexpr int NT = (N + 15) / 16;    // tile rows/cols (13 or 2)
    constexpr int NR = NT * 16;          // padded rows (208 or 32)
    constexpr int CS = N + 1;            // odd stride -> conflict-free row & col sweeps

    __shared__ float  Cmat[N * CS];
    __shared__ __bf16 Ast[NR * AST];
    __shared__ __bf16 Bst[NR * AST];
    __shared__ float  invnX[N];
    __shared__ float  invnY[N];
    __shared__ float  uvec[N], vvec[N], svec[N], wvec[N], avec[N];
    __shared__ float  lsum;

    const int tid  = threadIdx.x;
    const int lane = tid & 31;
    const int wave = tid >> 5;
    const int b    = blockIdx.x;
    const float* Xb = X + (size_t)b * N * D_DIM;
    const float* Yb = Y + (size_t)b * N * D_DIM;

    // ---- phase 0: inverse row norms (one wave per row, coalesced) ----
    for (int r = wave; r < N; r += NT) {
        float sx = 0.f, sy = 0.f;
        const float* xr = Xb + r * D_DIM;
        const float* yr = Yb + r * D_DIM;
        for (int k = lane; k < D_DIM; k += 32) {
            float xv = xr[k]; sx += xv * xv;
            float yv = yr[k]; sy += yv * yv;
        }
        #pragma unroll
        for (int off = 16; off > 0; off >>= 1) {
            sx += __shfl_xor(sx, off, 32);
            sy += __shfl_xor(sy, off, 32);
        }
        if (lane == 0) {
            invnX[r] = 1.0f / (sqrtf(sx) + 1e-12f);
            invnY[r] = 1.0f / (sqrtf(sy) + 1e-12f);
        }
    }
    if (tid == 0) lsum = 0.0f;

    // ---- phase 1: C = 1 - Xn * Yn^T via v_wmma_f32_16x16x32_bf16 ----
    v8f acc[NT];
    #pragma unroll
    for (int t = 0; t < NT; ++t) {
        #pragma unroll
        for (int e = 0; e < 8; ++e) acc[t][e] = 0.0f;
    }

    const int ti   = wave;                         // tile-row owned by this wave
    const int srow = tid >> 1;                     // staging: 2 threads per row
    const int koff = (tid & 1) * 16;               // 16 elems each
    const int crow = (srow < N) ? srow : (N - 1);  // clamp padded rows

    for (int kc = 0; kc < D_DIM; kc += KC) {
        __syncthreads();   // staging buffers free (also fences phase-0 writes)
        {
            const float* xs = Xb + crow * D_DIM + kc + koff;
            const float* ys = Yb + crow * D_DIM + kc + koff;
            const float sx = invnX[crow], sy = invnY[crow];
            __bf16* ad = Ast + srow * AST + koff;
            __bf16* bd = Bst + srow * AST + koff;
            v8bf pa0, pa1, pb0, pb1;
            #pragma unroll
            for (int q = 0; q < 8; ++q) {
                pa0[q] = (__bf16)(xs[q]     * sx);
                pa1[q] = (__bf16)(xs[q + 8] * sx);
                pb0[q] = (__bf16)(ys[q]     * sy);
                pb1[q] = (__bf16)(ys[q + 8] * sy);
            }
            *(v8bf*)(ad)     = pa0;  *(v8bf*)(ad + 8) = pa1;   // ds_store_b128 x2
            *(v8bf*)(bd)     = pb0;  *(v8bf*)(bd + 8) = pb1;
        }
        __syncthreads();

        const v16bf af = load_fragA(Ast, ti, lane);
        #pragma unroll
        for (int tj = 0; tj < NT; ++tj) {
            const v16bf bf = load_fragB(Bst, tj, lane);
            acc[tj] = __builtin_amdgcn_wmma_f32_16x16x32_bf16(
                false, af, false, bf, (short)0, acc[tj], false, false);
        }
    }

    // store C tiles (C/D layout: lane half -> M band, VGPR v -> row within band)
    {
        const int hl   = lane >> 4;
        const int col0 = lane & 15;
        #pragma unroll
        for (int tj = 0; tj < NT; ++tj) {
            const int col = tj * 16 + col0;
            if (col < N) {
                #pragma unroll
                for (int v = 0; v < 8; ++v) {
                    const int m = ti * 16 + hl * 8 + v;
                    if (m < N) Cmat[m * CS + col] = 1.0f - acc[tj][v];
                }
            }
        }
    }
    __syncthreads();

    // ---- phase 2: IPOT, matrix-free via T_k = diag(u) exp(-2k C) diag(v) ----
    const float MIU = 1.0f / (float)N;   // == nu (n == m)
    if (tid < N) { uvec[tid] = 1.0f; vvec[tid] = 1.0f; svec[tid] = MIU; }
    __syncthreads();

    for (int k = 1; k <= IPOT_ITERS; ++k) {
        const float ce = -2.0f * (float)k;
        if (tid < N) wvec[tid] = vvec[tid] * svec[tid];
        __syncthreads();
        if (tid < N) {                          // row pass: delta
            const float* row = Cmat + tid * CS;
            float s = 0.f;
            for (int j = 0; j < N; ++j) s += __expf(ce * row[j]) * wvec[j];
            const float dlt = MIU / (uvec[tid] * s + 1e-6f);
            avec[tid] = uvec[tid] * dlt;        // u_old * delta
        }
        __syncthreads();
        if (tid < N) {                          // col pass: sigma
            float t = 0.f;
            for (int i = 0; i < N; ++i) t += __expf(ce * Cmat[i * CS + tid]) * avec[i];
            const float sn = MIU / (vvec[tid] * t + 1e-6f);
            svec[tid]  = sn;
            vvec[tid] *= sn;
            uvec[tid]  = avec[tid];
        }
        __syncthreads();
    }

    // ---- phase 3: loss_b = -sum u_i exp(-2*ITERS*C) v_j C_ij ----
    if (tid < N) {
        const float* row = Cmat + tid * CS;
        const float ce = -2.0f * (float)IPOT_ITERS;
        float s = 0.f;
        for (int j = 0; j < N; ++j) {
            const float c = row[j];
            s += __expf(ce * c) * vvec[j] * c;
        }
        atomicAdd(&lsum, -(uvec[tid] * s));     // ds_add_f32
    }
    __syncthreads();
    if (tid == 0) partial[b] = lsum;
}

__global__ __launch_bounds__(128)
void finalize_kernel(const float* __restrict__ psp,
                     const float* __restrict__ ptm,
                     float* __restrict__ out) {
    __shared__ float sm[128];
    const int tid = threadIdx.x;
    sm[tid] = psp[tid] + ptm[tid];
    __syncthreads();
    #pragma unroll
    for (int off = 64; off > 0; off >>= 1) {
        if (tid < off) sm[tid] += sm[tid + off];
        __syncthreads();
    }
    if (tid == 0) out[0] = sm[0] * (1.0f / 128.0f);
}

extern "C" void kernel_launch(void* const* d_in, const int* in_sizes, int n_in,
                              void* d_out, int out_size, void* d_ws, size_t ws_size,
                              hipStream_t stream) {
    const float* sp1 = (const float*)d_in[0];  // [128,196,768]
    const float* sp2 = (const float*)d_in[1];
    const float* tm1 = (const float*)d_in[2];  // [128,32,768]
    const float* tm2 = (const float*)d_in[3];

    float* psp = (float*)d_ws;        // 128 per-batch spatial losses
    float* ptm = psp + 128;           // 128 per-batch temporal losses

    align_ipot_kernel<196><<<128, 13 * 32, 0, stream>>>(sp1, sp2, psp);
    align_ipot_kernel< 32><<<128,  2 * 32, 0, stream>>>(tm1, tm2, ptm);
    finalize_kernel<<<1, 128, 0, stream>>>(psp, ptm, (float*)d_out);
}
